// GCNGraphClassifier_77704548319503
// MI455X (gfx1250) — matrix operand order repaired and checked by reading, not compiled, not run on hardware
//
#include <hip/hip_runtime.h>
#include <hip/hip_fp16.h>

#define N_NODES  50000
#define N_EDGES  800000
#define N_GRAPHS 500
#define IN_DIM   128
#define HID      64
#define OUT_DIM  10

typedef __attribute__((ext_vector_type(16))) _Float16 v16h;
typedef __attribute__((ext_vector_type(8)))  float    v8f;

// ---------------- degree / norm ----------------
__global__ void k_init_deg(float* deg) {
    int i = blockIdx.x * blockDim.x + threadIdx.x;
    if (i < N_NODES) deg[i] = 1.0f;               // self-loop contributes 1
}

__global__ void k_deg_accum(const int* __restrict__ dst, float* deg) {
    int e = blockIdx.x * blockDim.x + threadIdx.x;
    if (e < N_EDGES) unsafeAtomicAdd(&deg[dst[e]], 1.0f);
}

__global__ void k_deg_rsqrt(float* deg) {
    int i = blockIdx.x * blockDim.x + threadIdx.x;
    if (i < N_NODES) deg[i] = rsqrtf(deg[i]);     // deg >= 1 always
}

// ---- WMMA GEMM + fused epilogue ----
// lin[M,64] = act(X[M,K]) @ W[K,64]
// out[M,64] = bias + lin * dis[m]^2            (self-loop message + bias init)
// One wave32 owns a 16-row M strip and computes all 4 N-tiles, loading the
// A operand once per 32-wide K chunk (4 v_wmma per chunk, A reused).
__global__ void k_gemm_wmma(const float* __restrict__ X, const float* __restrict__ W,
                            const float* __restrict__ bias, const float* __restrict__ dis,
                            float* __restrict__ lin, float* __restrict__ out,
                            int M, int K, int relu_in) {
    const int wavesPerBlock = blockDim.x >> 5;
    int wave = blockIdx.x * wavesPerBlock + (threadIdx.x >> 5);
    int lane = threadIdx.x & 31;
    if (wave * 16 >= M) return;

    int group = lane >> 4;                        // 0 or 1
    int mrow  = wave * 16 + (lane & 15);          // A: one M row per lane
    const float* __restrict__ xrow = X + (long)mrow * K;

    v8f acc0 = {}, acc1 = {}, acc2 = {}, acc3 = {};
    for (int k0 = 0; k0 < K; k0 += 32) {
        // ---- A operand: 16x32 f16; per-lane K ordering per CDNA5 ISA ----
        v16h a;
#pragma unroll
        for (int p = 0; p < 8; ++p) {
            int kk = k0 + ((p >> 2) << 4) + (group << 3) + ((p & 3) << 1);
            float a0 = xrow[kk];
            float a1 = xrow[kk + 1];
            if (relu_in) { a0 = fmaxf(a0, 0.0f); a1 = fmaxf(a1, 0.0f); }
            a[2 * p]     = (_Float16)a0;
            a[2 * p + 1] = (_Float16)a1;
        }
        // ---- B operands: lane holds N-column (lane&15) of W, K in A's order; 4 N-tiles ----
        const float* __restrict__ wrow = W + (long)k0 * HID + (lane & 15);
#pragma unroll
        for (int t = 0; t < 4; ++t) {
            v16h b;
#pragma unroll
            for (int p = 0; p < 8; ++p) {
                b[2 * p]     = (_Float16)wrow[(((p >> 2) << 4) + (group << 3) + ((p & 3) << 1)) * HID + t * 16];
                b[2 * p + 1] = (_Float16)wrow[(((p >> 2) << 4) + (group << 3) + ((p & 3) << 1) + 1) * HID + t * 16];
            }
            v8f c = (t == 0) ? acc0 : (t == 1) ? acc1 : (t == 2) ? acc2 : acc3;
            c = __builtin_amdgcn_wmma_f32_16x16x32_f16(
                false, a, false, b, (short)0, c, false, false);
            if      (t == 0) acc0 = c;
            else if (t == 1) acc1 = c;
            else if (t == 2) acc2 = c;
            else             acc3 = c;
        }
    }

    // ---- fused epilogue: lane holds N = t*16 + (lane&15); VGPR v -> M row ----
    float d2[8];
#pragma unroll
    for (int v = 0; v < 8; ++v) {
        float d = dis[wave * 16 + v + 8 * group];
        d2[v] = d * d;
    }
#pragma unroll
    for (int t = 0; t < 4; ++t) {
        v8f c = (t == 0) ? acc0 : (t == 1) ? acc1 : (t == 2) ? acc2 : acc3;
        int n = t * 16 + (lane & 15);
        float bn = bias[n];
#pragma unroll
        for (int v = 0; v < 8; ++v) {
            long idx = (long)(wave * 16 + v + 8 * group) * HID + n;
            lin[idx] = c[v];
            out[idx] = bn + c[v] * d2[v];
        }
    }
}

// ---------------- edge scatter: 16 threads per edge, float4 per thread ----------------
__global__ void k_scatter(const float* __restrict__ lin, const int* __restrict__ src,
                          const int* __restrict__ dst, const float* __restrict__ dis,
                          float* __restrict__ out) {
    int gid = blockIdx.x * blockDim.x + threadIdx.x;
    int e = gid >> 4;
    if (e >= N_EDGES) return;
    int c = (gid & 15) << 2;
    int s = src[e], d = dst[e];
    float norm = dis[s] * dis[d];
    const float4 m = *(const float4*)(lin + (long)s * HID + c);
    float* o = out + (long)d * HID + c;
    unsafeAtomicAdd(o + 0, m.x * norm);
    unsafeAtomicAdd(o + 1, m.y * norm);
    unsafeAtomicAdd(o + 2, m.z * norm);
    unsafeAtomicAdd(o + 3, m.w * norm);
}

// ---------------- pooling ----------------
__global__ void k_zero(float* p, int n) {
    int i = blockIdx.x * blockDim.x + threadIdx.x;
    if (i < n) p[i] = 0.0f;
}

__global__ void k_pool_accum(const float* __restrict__ h, const int* __restrict__ batch,
                             float* pooled) {
    int i = blockIdx.x * blockDim.x + threadIdx.x;
    if (i < N_NODES * HID) {
        int n = i >> 6, f = i & 63;
        unsafeAtomicAdd(&pooled[batch[n] * HID + f], fmaxf(h[i], 0.0f));  // ReLU fused
    }
}

__global__ void k_pool_counts(const int* __restrict__ batch, float* counts) {
    int n = blockIdx.x * blockDim.x + threadIdx.x;
    if (n < N_NODES) unsafeAtomicAdd(&counts[batch[n]], 1.0f);
}

__global__ void k_pool_div(float* pooled, const float* __restrict__ counts) {
    int i = blockIdx.x * blockDim.x + threadIdx.x;
    if (i < N_GRAPHS * HID) pooled[i] /= fmaxf(counts[i >> 6], 1.0f);
}

// ---------------- MLP head: one block (64 threads) per graph ----------------
__global__ void k_mlp(const float* __restrict__ pooled,
                      const float* __restrict__ W1, const float* __restrict__ b1,
                      const float* __restrict__ W2, const float* __restrict__ b2,
                      float* __restrict__ out) {
    __shared__ float pin[HID];
    __shared__ float hid[HID];
    int g = blockIdx.x, t = threadIdx.x;
    pin[t] = pooled[g * HID + t];
    __syncthreads();
    float s = b1[t];
#pragma unroll 8
    for (int k = 0; k < HID; ++k) s += pin[k] * W1[k * HID + t];
    hid[t] = fmaxf(s, 0.0f);
    __syncthreads();
    if (t < OUT_DIM) {
        float o = b2[t];
#pragma unroll 8
        for (int k = 0; k < HID; ++k) o += hid[k] * W2[k * OUT_DIM + t];
        out[g * OUT_DIM + t] = o;
    }
}

__global__ void k_tail(float* out) {
    if (blockIdx.x == 0 && threadIdx.x == 0) out[N_GRAPHS * OUT_DIM] = 0.0f;  // aux scalar
}

extern "C" void kernel_launch(void* const* d_in, const int* in_sizes, int n_in,
                              void* d_out, int out_size, void* d_ws, size_t ws_size,
                              hipStream_t stream) {
    // dict order: x, edge_index, batch, W1,b1, W2,b2, W3,b3, lin_W1,lin_b1, lin_W2,lin_b2
    const float* x     = (const float*)d_in[0];
    const int*   eidx  = (const int*)d_in[1];
    const int*   batch = (const int*)d_in[2];
    const float* W1 = (const float*)d_in[3];  const float* b1 = (const float*)d_in[4];
    const float* W2 = (const float*)d_in[5];  const float* b2 = (const float*)d_in[6];
    const float* W3 = (const float*)d_in[7];  const float* b3 = (const float*)d_in[8];
    const float* lW1 = (const float*)d_in[9];  const float* lb1 = (const float*)d_in[10];
    const float* lW2 = (const float*)d_in[11]; const float* lb2 = (const float*)d_in[12];
    float* out = (float*)d_out;

    const int* src = eidx;            // edge_index[0]
    const int* dst = eidx + N_EDGES;  // edge_index[1]

    // workspace layout (floats)
    float* ws     = (float*)d_ws;
    float* lin    = ws;                         // N_NODES*HID
    float* bufA   = lin  + (size_t)N_NODES * HID;
    float* bufB   = bufA + (size_t)N_NODES * HID;
    float* dis    = bufB + (size_t)N_NODES * HID;  // N_NODES (deg -> rsqrt in place)
    float* pooled = dis  + N_NODES;                // N_GRAPHS*HID
    float* counts = pooled + (size_t)N_GRAPHS * HID;  // N_GRAPHS

    const int B = 256;
    const int nodeB = (N_NODES + B - 1) / B;
    const int edgeB = (N_EDGES + B - 1) / B;
    const int nhB   = (N_NODES * HID + B - 1) / B;
    const int gemmWaves  = N_NODES / 16;                    // 3125 M-strips
    const int gemmBlocks = (gemmWaves + (B / 32) - 1) / (B / 32);
    const int scatB = (N_EDGES * 16 + B - 1) / B;

    // degrees -> deg_inv_sqrt
    hipLaunchKernelGGL(k_init_deg,  dim3(nodeB), dim3(B), 0, stream, dis);
    hipLaunchKernelGGL(k_deg_accum, dim3(edgeB), dim3(B), 0, stream, dst, dis);
    hipLaunchKernelGGL(k_deg_rsqrt, dim3(nodeB), dim3(B), 0, stream, dis);

    // layer 1: x(IN_DIM) -> bufA
    hipLaunchKernelGGL(k_gemm_wmma, dim3(gemmBlocks), dim3(B), 0, stream,
                       x, W1, b1, dis, lin, bufA, N_NODES, IN_DIM, 0);
    hipLaunchKernelGGL(k_scatter,   dim3(scatB), dim3(B), 0, stream, lin, src, dst, dis, bufA);

    // layer 2: relu(bufA) -> bufB
    hipLaunchKernelGGL(k_gemm_wmma, dim3(gemmBlocks), dim3(B), 0, stream,
                       bufA, W2, b2, dis, lin, bufB, N_NODES, HID, 1);
    hipLaunchKernelGGL(k_scatter,   dim3(scatB), dim3(B), 0, stream, lin, src, dst, dis, bufB);

    // layer 3: relu(bufB) -> bufA
    hipLaunchKernelGGL(k_gemm_wmma, dim3(gemmBlocks), dim3(B), 0, stream,
                       bufB, W3, b3, dis, lin, bufA, N_NODES, HID, 1);
    hipLaunchKernelGGL(k_scatter,   dim3(scatB), dim3(B), 0, stream, lin, src, dst, dis, bufA);

    // mean pool (relu fused into accumulation)
    const int poolN = N_GRAPHS * HID + N_GRAPHS;
    hipLaunchKernelGGL(k_zero, dim3((poolN + B - 1) / B), dim3(B), 0, stream, pooled, poolN);
    hipLaunchKernelGGL(k_pool_accum,  dim3(nhB),   dim3(B), 0, stream, bufA, batch, pooled);
    hipLaunchKernelGGL(k_pool_counts, dim3(nodeB), dim3(B), 0, stream, batch, counts);
    hipLaunchKernelGGL(k_pool_div, dim3((N_GRAPHS * HID + B - 1) / B), dim3(B), 0, stream,
                       pooled, counts);

    // MLP head -> logits, plus trailing aux zero
    hipLaunchKernelGGL(k_mlp, dim3(N_GRAPHS), dim3(HID), 0, stream,
                       pooled, lW1, lb1, lW2, lb2, out);
    hipLaunchKernelGGL(k_tail, dim3(1), dim3(1), 0, stream, out);
}
